// SOZClassifier_5677946765443
// MI455X (gfx1250) — matrix operand order repaired and checked by reading, not compiled, not run on hardware
//
#include <hip/hip_runtime.h>
#include <hip/hip_bf16.h>

// ---------------------------------------------------------------------------
// GraphSAGE (3x SAGEConv + BN + ReLU) + MLP head for MI455X (gfx1250).
// Matrix ops via v_wmma_f32_16x16x32_bf16 (wave32 WMMA), 2 M-tiles/wave
// register blocking, global_prefetch_b8 latency hiding on the edge gather.
// ---------------------------------------------------------------------------

#define N_NODES 100000
#define N_EDGES 3200000
#define IN_CH   128
#define HID     256
#define BN_EPS  1e-5f

typedef __attribute__((ext_vector_type(16))) __bf16 v16bf;
typedef __attribute__((ext_vector_type(8)))  __bf16 v8bf;
typedef __attribute__((ext_vector_type(8)))  float  v8f;

// ---------------------------------------------------------------------------
// Utility kernels
// ---------------------------------------------------------------------------

__global__ void k_zero_f32(float* __restrict__ p, long long n) {
    long long i = (long long)blockIdx.x * blockDim.x + threadIdx.x;
    long long stride = (long long)gridDim.x * blockDim.x;
    for (; i < n; i += stride) p[i] = 0.0f;
}

// degree: deg[dst[e]] += 1
__global__ void k_degree(const long long* __restrict__ dst, float* __restrict__ deg, int nE) {
    int e = blockIdx.x * blockDim.x + threadIdx.x;
    if (e < nE) atomicAdd(&deg[(int)dst[e]], 1.0f);
}

// scatter-add node features along edges: agg[dst] += h[src]
// 4 edges per block; prefetch next edge's source row to hide L2 latency.
#define EDGES_PER_BLOCK 4
__global__ void k_scatter(const float* __restrict__ h,
                          const long long* __restrict__ src,
                          const long long* __restrict__ dst,
                          float* __restrict__ agg, int K) {
    int c  = threadIdx.x;                       // blockDim.x == K
    int e0 = blockIdx.x * EDGES_PER_BLOCK;
    int s_next = (int)src[e0];
    __builtin_prefetch(&h[(size_t)s_next * K + c], 0, 3);
#pragma unroll
    for (int i = 0; i < EDGES_PER_BLOCK; ++i) {
        int e = e0 + i;
        int s = s_next;
        int d = (int)dst[e];
        if (i + 1 < EDGES_PER_BLOCK) {
            s_next = (int)src[e + 1];
            __builtin_prefetch(&h[(size_t)s_next * K + c], 0, 3);
        }
        float v = h[(size_t)s * K + c];
        atomicAdd(&agg[(size_t)d * K + c], v);
    }
}

// mean in-place: agg[n][c] /= max(deg[n], 1)
__global__ void k_mean_inplace(float* __restrict__ agg, const float* __restrict__ deg,
                               long long total, int K) {
    long long i = (long long)blockIdx.x * blockDim.x + threadIdx.x;
    long long stride = (long long)gridDim.x * blockDim.x;
    for (; i < total; i += stride) {
        int n = (int)(i / K);
        float d = deg[n];
        agg[i] = agg[i] / fmaxf(d, 1.0f);
    }
}

// weight prep: W (K x Nout, row-major f32) -> Wt (Nout x K, row-major bf16)
__global__ void k_wprep(const float* __restrict__ W, __bf16* __restrict__ Wt,
                        int K, int Nout) {
    int i = blockIdx.x * blockDim.x + threadIdx.x;
    int total = K * Nout;
    if (i >= total) return;
    int n = i / K;
    int k = i - n * K;
    Wt[(size_t)n * K + k] = (__bf16)W[(size_t)k * Nout + n];
}

// ---------------------------------------------------------------------------
// WMMA GEMM: out[M x Nout] = A1[M x K1] @ B1t^T + A2[M x K2] @ B2t^T + bias
// A* are f32 row-major (converted in-register to bf16); B*t are bf16 N-major.
// Each wave computes TWO 16x16 D tiles (32 rows x 16 cols), reusing each B
// fragment for two independent WMMAs (better XDL pipelining, half B traffic).
// M = N_NODES = 3125 * 32 exactly.
// ---------------------------------------------------------------------------

// A fragment, 16-bit 16x32 layout:
//   lanes 0-15 : row M=lane,    elems K = k0+{0..7}  then k0+{16..23}
//   lanes 16-31: row M=lane-16, elems K = k0+{8..15} then k0+{24..31}
__device__ __forceinline__ v16bf load_a_f32(const float* __restrict__ A,
                                            int row, int ldk, int k0, int half) {
    const float* p = A + (size_t)row * ldk + k0 + half * 8;
    float4 f0 = *(const float4*)(p);
    float4 f1 = *(const float4*)(p + 4);
    float4 f2 = *(const float4*)(p + 16);
    float4 f3 = *(const float4*)(p + 20);
    v16bf a;
    a[0]  = (__bf16)f0.x; a[1]  = (__bf16)f0.y; a[2]  = (__bf16)f0.z; a[3]  = (__bf16)f0.w;
    a[4]  = (__bf16)f1.x; a[5]  = (__bf16)f1.y; a[6]  = (__bf16)f1.z; a[7]  = (__bf16)f1.w;
    a[8]  = (__bf16)f2.x; a[9]  = (__bf16)f2.y; a[10] = (__bf16)f2.z; a[11] = (__bf16)f2.w;
    a[12] = (__bf16)f3.x; a[13] = (__bf16)f3.y; a[14] = (__bf16)f3.z; a[15] = (__bf16)f3.w;
    return a;
}

// B fragment, 16-bit 32x16 layout (from Bt stored N-major):
//   lanes 0-15 : col N=lane,    elems K = k0+{0..7}  then k0+{16..23}
//   lanes 16-31: col N=lane-16, elems K = k0+{8..15} then k0+{24..31}
__device__ __forceinline__ v16bf load_b_bf(const __bf16* __restrict__ Bt,
                                           int col, int ldk, int k0, int half) {
    const __bf16* p = Bt + (size_t)col * ldk + k0 + half * 8;
    v8bf b0 = *(const v8bf*)(p);
    v8bf b1 = *(const v8bf*)(p + 16);
    return __builtin_shufflevector(b0, b1, 0, 1, 2, 3, 4, 5, 6, 7,
                                            8, 9, 10, 11, 12, 13, 14, 15);
}

#define GEMM_WAVES 4   // waves per block, each owns a distinct 16-col tile
#define GEMM_MT    2   // M-tiles (16 rows each) per wave

__global__ void k_gemm_wmma(const float* __restrict__ A1, const __bf16* __restrict__ B1t, int K1,
                            const float* __restrict__ A2, const __bf16* __restrict__ B2t, int K2,
                            const float* __restrict__ bias, float* __restrict__ out,
                            int Nout, int relu) {
    const int wave  = threadIdx.x >> 5;
    const int lane  = threadIdx.x & 31;
    const int half  = lane >> 4;
    const int lrow  = lane & 15;
    const int mT0   = blockIdx.x * GEMM_MT;
    const int nTile = blockIdx.y * GEMM_WAVES + wave;
    const int row0  = mT0 * 16 + lrow;     // A row, first M tile
    const int row1  = row0 + 16;           // A row, second M tile
    const int col   = nTile * 16 + lrow;   // B col / D col for this lane

    v8f acc0 = {}, acc1 = {};
    for (int k0 = 0; k0 < K1; k0 += 32) {
        if (k0 + 32 < K1) {
            __builtin_prefetch(A1 + (size_t)row0 * K1 + k0 + 32, 0, 3);
            __builtin_prefetch(A1 + (size_t)row1 * K1 + k0 + 32, 0, 3);
        }
        v16bf b  = load_b_bf(B1t, col, K1, k0, half);
        v16bf a0 = load_a_f32(A1, row0, K1, k0, half);
        v16bf a1 = load_a_f32(A1, row1, K1, k0, half);
        acc0 = __builtin_amdgcn_wmma_f32_16x16x32_bf16(
            false, a0, false, b, (short)0, acc0, false, false);
        acc1 = __builtin_amdgcn_wmma_f32_16x16x32_bf16(
            false, a1, false, b, (short)0, acc1, false, false);
    }
    if (A2) {
        for (int k0 = 0; k0 < K2; k0 += 32) {
            if (k0 + 32 < K2) {
                __builtin_prefetch(A2 + (size_t)row0 * K2 + k0 + 32, 0, 3);
                __builtin_prefetch(A2 + (size_t)row1 * K2 + k0 + 32, 0, 3);
            }
            v16bf b  = load_b_bf(B2t, col, K2, k0, half);
            v16bf a0 = load_a_f32(A2, row0, K2, k0, half);
            v16bf a1 = load_a_f32(A2, row1, K2, k0, half);
            acc0 = __builtin_amdgcn_wmma_f32_16x16x32_bf16(
                false, a0, false, b, (short)0, acc0, false, false);
            acc1 = __builtin_amdgcn_wmma_f32_16x16x32_bf16(
                false, a1, false, b, (short)0, acc1, false, false);
        }
    }

    const float bv = bias ? bias[col] : 0.0f;
    // D layout: lanes 0-15 -> M rows 0..7 of tile, lanes 16-31 -> rows 8..15
    const int mbase0 = mT0 * 16 + half * 8;
    const int mbase1 = mbase0 + 16;
#pragma unroll
    for (int r = 0; r < 8; ++r) {
        float v0 = acc0[r] + bv;
        float v1 = acc1[r] + bv;
        if (relu) { v0 = fmaxf(v0, 0.0f); v1 = fmaxf(v1, 0.0f); }
        out[(size_t)(mbase0 + r) * Nout + col] = v0;
        out[(size_t)(mbase1 + r) * Nout + col] = v1;
    }
}

// ---------------------------------------------------------------------------
// BatchNorm (two-pass): column sums/sumsq then normalize(+ReLU) in place.
// ---------------------------------------------------------------------------

#define STATS_ROWS 256

__global__ void k_stats(const float* __restrict__ h, float* __restrict__ stats,
                        int N, int Nout) {
    int c = threadIdx.x;                 // blockDim.x == Nout
    int r0 = blockIdx.x * STATS_ROWS;
    int r1 = min(r0 + STATS_ROWS, N);
    float s = 0.0f, s2 = 0.0f;
    for (int r = r0; r < r1; ++r) {
        float v = h[(size_t)r * Nout + c];   // fully coalesced across threads
        s += v; s2 += v * v;
    }
    atomicAdd(&stats[c], s);
    atomicAdd(&stats[Nout + c], s2);
}

__global__ void k_bnorm(float* __restrict__ h, const float* __restrict__ stats,
                        const float* __restrict__ gamma, const float* __restrict__ beta,
                        long long total, int N, int Nout, int relu) {
    long long i = (long long)blockIdx.x * blockDim.x + threadIdx.x;
    long long stride = (long long)gridDim.x * blockDim.x;
    float invN = 1.0f / (float)N;
    for (; i < total; i += stride) {
        int c = (int)(i % Nout);
        float m = stats[c] * invN;
        float v = stats[Nout + c] * invN - m * m;
        float y = gamma[c] * (h[i] - m) * rsqrtf(v + BN_EPS) + beta[c];
        if (relu) y = fmaxf(y, 0.0f);
        h[i] = y;
    }
}

// final head: out[n] = z[n,:] . W2 + c2
__global__ void k_mlp2(const float* __restrict__ z, const float* __restrict__ W2,
                       const float* __restrict__ c2, float* __restrict__ out, int N) {
    int n = blockIdx.x * blockDim.x + threadIdx.x;
    if (n >= N) return;
    float acc = 0.0f;
#pragma unroll 8
    for (int j = 0; j < HID / 2; ++j) acc += z[(size_t)n * (HID / 2) + j] * W2[j];
    out[n] = acc + c2[0];
}

// ---------------------------------------------------------------------------
// Host launch
// ---------------------------------------------------------------------------

static inline int ceil_div(long long a, int b) { return (int)((a + b - 1) / b); }

extern "C" void kernel_launch(void* const* d_in, const int* in_sizes, int n_in,
                              void* d_out, int out_size, void* d_ws, size_t ws_size,
                              hipStream_t stream) {
    (void)in_sizes; (void)n_in; (void)out_size; (void)ws_size;

    const float*     x   = (const float*)d_in[0];
    const long long* ei  = (const long long*)d_in[1];
    const long long* src = ei;
    const long long* dst = ei + N_EDGES;

    const float* Wl[3] = { (const float*)d_in[2],  (const float*)d_in[7],  (const float*)d_in[12] };
    const float* Wr[3] = { (const float*)d_in[3],  (const float*)d_in[8],  (const float*)d_in[13] };
    const float* b [3] = { (const float*)d_in[4],  (const float*)d_in[9],  (const float*)d_in[14] };
    const float* g [3] = { (const float*)d_in[5],  (const float*)d_in[10], (const float*)d_in[15] };
    const float* be[3] = { (const float*)d_in[6],  (const float*)d_in[11], (const float*)d_in[16] };
    const float* W1 = (const float*)d_in[17];
    const float* c1 = (const float*)d_in[18];
    const float* W2 = (const float*)d_in[19];
    const float* c2 = (const float*)d_in[20];
    float* out = (float*)d_out;

    // -- workspace carve (256B aligned) --
    char* ws = (char*)d_ws;
    size_t off = 0;
    auto carve = [&](size_t bytes) -> void* {
        void* p = ws + off;
        off += (bytes + 255) & ~(size_t)255;
        return p;
    };
    float*  deg   = (float*)carve((size_t)N_NODES * 4);
    float*  agg   = (float*)carve((size_t)N_NODES * HID * 4);   // also reused as MLP z-buffer
    float*  bufA  = (float*)carve((size_t)N_NODES * HID * 4);
    float*  bufB  = (float*)carve((size_t)N_NODES * HID * 4);
    __bf16* Wlt   = (__bf16*)carve((size_t)HID * HID * 2);
    __bf16* Wrt   = (__bf16*)carve((size_t)HID * HID * 2);
    float*  stats = (float*)carve(2 * HID * 4);

    const int ZB = 256;
    const int zgrid = 2048;  // grid-stride zero/elementwise kernels

    // degree (graph is static across layers; compute once)
    k_zero_f32<<<ceil_div(N_NODES, ZB), ZB, 0, stream>>>(deg, N_NODES);
    k_degree<<<ceil_div(N_EDGES, ZB), ZB, 0, stream>>>(dst, deg, N_EDGES);

    const float* h_in = x;
    float* h_out = bufA;
    int Kin = IN_CH;

    for (int l = 0; l < 3; ++l) {
        long long aggN = (long long)N_NODES * Kin;
        // mean aggregation
        k_zero_f32<<<zgrid, ZB, 0, stream>>>(agg, aggN);
        k_scatter<<<N_EDGES / EDGES_PER_BLOCK, Kin, 0, stream>>>(h_in, src, dst, agg, Kin);
        k_mean_inplace<<<zgrid, ZB, 0, stream>>>(agg, deg, aggN, Kin);
        // weights -> bf16 transposed
        k_wprep<<<ceil_div((long long)Kin * HID, ZB), ZB, 0, stream>>>(Wl[l], Wlt, Kin, HID);
        k_wprep<<<ceil_div((long long)Kin * HID, ZB), ZB, 0, stream>>>(Wr[l], Wrt, Kin, HID);
        // fused dual GEMM: mean@Wl + h@Wr + b
        dim3 ggrid(N_NODES / (16 * GEMM_MT), HID / (16 * GEMM_WAVES));
        k_gemm_wmma<<<ggrid, 32 * GEMM_WAVES, 0, stream>>>(
            agg, Wlt, Kin, h_in, Wrt, Kin, b[l], h_out, HID, /*relu=*/0);
        // batch norm (+ ReLU except last conv layer)
        k_zero_f32<<<ceil_div(2 * HID, ZB), ZB, 0, stream>>>(stats, 2 * HID);
        k_stats<<<ceil_div(N_NODES, STATS_ROWS), HID, 0, stream>>>(h_out, stats, N_NODES, HID);
        k_bnorm<<<zgrid, ZB, 0, stream>>>(h_out, stats, g[l], be[l],
                                          (long long)N_NODES * HID, N_NODES, HID,
                                          /*relu=*/(l < 2) ? 1 : 0);
        // rotate buffers
        h_in = h_out;
        h_out = (h_out == bufA) ? bufB : bufA;
        Kin = HID;
    }

    // MLP head: z = relu(h @ W1 + c1)   (W1: 256 x 128)
    float* z = agg;  // agg no longer needed; reuse as N x 128 buffer
    k_wprep<<<ceil_div((long long)HID * (HID / 2), ZB), ZB, 0, stream>>>(W1, Wlt, HID, HID / 2);
    dim3 mgrid(N_NODES / (16 * GEMM_MT), (HID / 2) / (16 * GEMM_WAVES));
    k_gemm_wmma<<<mgrid, 32 * GEMM_WAVES, 0, stream>>>(
        h_in, Wlt, HID, nullptr, nullptr, 0, c1, z, HID / 2, /*relu=*/1);

    // out = z @ W2 + c2
    k_mlp2<<<ceil_div(N_NODES, ZB), ZB, 0, stream>>>(z, W2, c2, out, N_NODES);
}
